// DualAttention_41016937677188
// MI455X (gfx1250) — matrix operand order repaired and checked by reading, not compiled, not run on hardware
//
#include <hip/hip_runtime.h>
#include <hip/hip_bf16.h>

typedef _Float16 half8 __attribute__((ext_vector_type(8)));
typedef _Float16 v16h  __attribute__((ext_vector_type(16)));
typedef float    v8f   __attribute__((ext_vector_type(8)));

// ---------------------------------------------------------------------------
// WMMA helpers (CDNA5 wave32, v_wmma_f32_16x16x32_f16)
// A 16x32 f16: lanes 0-15 -> row=lane, K {0..7,16..23}; lanes 16-31 -> K {8..15,24..31}
// B 32x16 f16: lanes 0-15 -> col=lane, K {0..7,16..23}; lanes 16-31 -> K {8..15,24..31}
// C/D 16x16 f32: vgpr r -> row r (lanes 0-15) / r+8 (lanes 16-31); col = lane&15
// ---------------------------------------------------------------------------
__device__ __forceinline__ v8f wmma_f16(v16h a, v16h b, v8f c) {
    return __builtin_amdgcn_wmma_f32_16x16x32_f16(false, a, false, b, (short)0, c,
                                                  false, false);
}

// element (r,k) at base[(row+r)*stride + k], 16 rows x 32 k (works on LDS or global)
__device__ __forceinline__ v16h load_fragA_k32(const _Float16* base, int row, int stride) {
    int lane = threadIdx.x & 31;
    const _Float16* p = base + (size_t)(row + (lane & 15)) * stride + ((lane >> 4) * 8);
    union { v16h v; half8 h[2]; } u;
    u.h[0] = *(const half8*)p;
    u.h[1] = *(const half8*)(p + 16);
    return u.v;
}

// A fragment with only K=0..15 valid (upper K zero)
__device__ __forceinline__ v16h load_fragA_k16(const _Float16* base, int row, int stride) {
    int lane = threadIdx.x & 31;
    const _Float16* p = base + (size_t)(row + (lane & 15)) * stride + ((lane >> 4) * 8);
    union { v16h v; half8 h[2]; } u;
    half8 z = {};
    u.h[0] = *(const half8*)p;
    u.h[1] = z;
    return u.v;
}

// element (k,n) stored as base[(row+n)*stride + k]  (i.e. [n][k] row-major)
__device__ __forceinline__ v16h load_fragB_k32(const _Float16* base, int row, int stride) {
    int lane = threadIdx.x & 31;
    const _Float16* p = base + (size_t)(row + (lane & 15)) * stride + ((lane >> 4) * 8);
    union { v16h v; half8 h[2]; } u;
    u.h[0] = *(const half8*)p;
    u.h[1] = *(const half8*)(p + 16);
    return u.v;
}

// B fragment with only K=0..15 valid (upper K zero)
__device__ __forceinline__ v16h load_fragB_k16(const _Float16* base, int row, int stride) {
    int lane = threadIdx.x & 31;
    const _Float16* p = base + (size_t)(row + (lane & 15)) * stride + ((lane >> 4) * 8);
    union { v16h v; half8 h[2]; } u;
    half8 z = {};
    u.h[0] = *(const half8*)p;
    u.h[1] = z;
    return u.v;
}

#define LDSPAD 40  // halves per row (32 data + 8 pad) -> 80B rows, 16B aligned frags

// ---------------------------------------------------------------------------
// Repack kernels (run once per launch; tiny): weights -> f16, GEMM-native layout.
// ---------------------------------------------------------------------------
// conv3 weights [M=128][Cin][3] -> [3][128][Cin]  (A-frag direct-global layout)
__global__ __launch_bounds__(256)
void repack3_kernel(const float* __restrict__ w, _Float16* __restrict__ out, int Cin) {
    int flat = blockIdx.x * 256 + threadIdx.x;
    int n = 3 * 128 * Cin;
    if (flat >= n) return;
    int dk  = flat / (128 * Cin);
    int rem = flat - dk * 128 * Cin;
    int m   = rem / Cin;
    int ci  = rem - m * Cin;
    out[flat] = (_Float16)w[(size_t)(m * Cin + ci) * 3 + dk];
}

// plain f32 -> f16 copy (1x1 conv weights, row-major [M][K] already GEMM-native)
__global__ __launch_bounds__(256)
void copy_h_kernel(const float* __restrict__ in, _Float16* __restrict__ out, int n) {
    int i = blockIdx.x * 256 + threadIdx.x;
    if (i < n) out[i] = (_Float16)in[i];
}

// ---------------------------------------------------------------------------
// K1: conv1d k=3 pad=1 (Cin -> 128) + BatchNorm + ReLU.  GEMM: M=128, N=P-tile,
// K = 3*Cin (tap-major chunks of 32).  Block = 8 waves, tile 128M x 128N.
// A-fragments straight from repacked f16 global weights; only B staged in LDS.
// ---------------------------------------------------------------------------
__global__ __launch_bounds__(256)
void conv3_cbr_kernel(const float* __restrict__ x,      // [B][Cin][P]
                      const _Float16* __restrict__ wh,  // [3][128][Cin]
                      const float* __restrict__ gg, const float* __restrict__ bb,
                      const float* __restrict__ mm, const float* __restrict__ vv,
                      float* __restrict__ out,          // [B][128][P]
                      int Cin, int P) {
    int b  = blockIdx.z;
    int p0 = blockIdx.x * 128;
    const float* xb = x + (size_t)b * Cin * P;
    float* ob = out + (size_t)b * 128 * P;

    __shared__ __align__(16) _Float16 bS[128 * LDSPAD];

    int tid = threadIdx.x, wave = tid >> 5, lane = tid & 31;
    int m0 = wave * 16;
    v8f acc[8] = {};

    int cgrp = Cin >> 5;          // K chunks per tap
    int nch  = cgrp * 3;
    // staging coords (hoisted)
    int skl = tid >> 3;
    int sn0 = (tid & 7) * 16;
    for (int kc = 0; kc < nch; ++kc) {
        int dk = kc / cgrp;
        int cb = (kc - dk * cgrp) << 5;
        __syncthreads();
        { // stage B: x tile as [128n][32k], pad=1 boundary zeros only on edge tiles
            const float* xr = xb + (size_t)(cb + skl) * P;
            int pstart = p0 + sn0 + dk - 1;
            if (pstart >= 0 && pstart + 16 <= P) {
#pragma unroll
                for (int u = 0; u < 16; ++u)
                    bS[(sn0 + u) * LDSPAD + skl] = (_Float16)xr[pstart + u];
            } else {
#pragma unroll
                for (int u = 0; u < 16; ++u) {
                    int p = pstart + u;
                    float val = (p >= 0 && p < P) ? xr[p] : 0.f;
                    bS[(sn0 + u) * LDSPAD + skl] = (_Float16)val;
                }
            }
            if (kc + 1 < nch) // prefetch next chunk of x (global_prefetch_b8)
                __builtin_prefetch(xb + (size_t)(((kc + 1) % cgrp) << 5) * P +
                                       (size_t)skl * P + p0 + sn0, 0, 1);
        }
        __syncthreads();
        v16h af = load_fragA_k32(wh + (size_t)dk * 128 * Cin + cb, m0, Cin);
        v16h bf[8];
#pragma unroll
        for (int ns = 0; ns < 8; ++ns) bf[ns] = load_fragB_k32(bS, ns * 16, LDSPAD);
#pragma unroll
        for (int ns = 0; ns < 8; ++ns) acc[ns] = wmma_f16(af, bf[ns], acc[ns]);
    }

    int hi = lane >> 4, nl = lane & 15;
#pragma unroll
    for (int r = 0; r < 8; ++r) {
        int m = m0 + r + 8 * hi;
        float sc = gg[m] * rsqrtf(vv[m] + 1e-5f);
        float sh = bb[m] - mm[m] * sc;
#pragma unroll
        for (int ns = 0; ns < 8; ++ns) {
            float y = fmaxf(acc[ns][r] * sc + sh, 0.f);
            ob[(size_t)m * P + p0 + ns * 16 + nl] = y;
        }
    }
}

// ---------------------------------------------------------------------------
// K2: 1x1 conv (K=128) + bias, f16 output.  Tile 16M x 128N, one wave per 16N.
// ---------------------------------------------------------------------------
__global__ __launch_bounds__(256)
void gemm1x1_h_kernel(const float* __restrict__ in,     // [B][128][P]
                      const _Float16* __restrict__ wh,  // [M][128] f16
                      const float* __restrict__ bias,   // [M]
                      _Float16* __restrict__ out,       // [B][M][P]
                      int M, int P) {
    int b  = blockIdx.z;
    int m0 = blockIdx.y * 16;
    int p0 = blockIdx.x * 128;
    const float* ib = in + (size_t)b * 128 * P;
    _Float16* ob = out + (size_t)b * M * P;

    __shared__ __align__(16) _Float16 bS[128 * LDSPAD];

    int tid = threadIdx.x, wave = tid >> 5, lane = tid & 31;
    v8f acc = {};

    for (int kc = 0; kc < 4; ++kc) {
        int kb = kc * 32;
        __syncthreads();
        {
            int kl = tid >> 3;
            int n0 = (tid & 7) * 16;
            const float* xr = ib + (size_t)(kb + kl) * P + p0 + n0;
#pragma unroll
            for (int u = 0; u < 16; ++u)
                bS[(n0 + u) * LDSPAD + kl] = (_Float16)xr[u];
            if (kc < 3) __builtin_prefetch(xr + 32 * P, 0, 1);
        }
        __syncthreads();
        v16h af = load_fragA_k32(wh + (size_t)m0 * 128 + kb, 0, 128);
        v16h bf = load_fragB_k32(bS, wave * 16, LDSPAD);
        acc = wmma_f16(af, bf, acc);
    }

    int hi = lane >> 4, nl = lane & 15;
    int n0 = p0 + wave * 16;
#pragma unroll
    for (int r = 0; r < 8; ++r) {
        int m = m0 + r + 8 * hi;
        ob[(size_t)m * P + n0 + nl] = (_Float16)(acc[r] + bias[m]);
    }
}

// ---------------------------------------------------------------------------
// K3: position attention, flash style.  Per wave: 16 query rows (i), loop j in
// chunks of 32: S = q k  (d=16, K-padded WMMA), online softmax, O += P~ V^T.
// Epilogue: sa_feat = alpha * O/rowsum + feat1.
// ---------------------------------------------------------------------------
__global__ __launch_bounds__(256)
void pos_attn_kernel(const _Float16* __restrict__ qh, // [B][16][P]
                     const _Float16* __restrict__ kh, // [B][16][P]
                     const _Float16* __restrict__ vh, // [B][128][P]
                     const float* __restrict__ feat1, // [B][128][P]
                     const float* __restrict__ alpha,
                     float* __restrict__ out,         // [B][128][P]
                     int P) {
    int b = blockIdx.y;
    int tid = threadIdx.x, wave = tid >> 5, lane = tid & 31;
    int i0 = blockIdx.x * 128 + wave * 16;

    const _Float16* qb = qh + (size_t)b * 16 * P;
    const _Float16* kb = kh + (size_t)b * 16 * P;
    const _Float16* vb = vh + (size_t)b * 128 * P;

    __shared__ __align__(16) _Float16 kS[32 * 16];        // [j][d]
    __shared__ __align__(16) _Float16 vS[128 * LDSPAD];   // [c][j]
    __shared__ __align__(16) _Float16 pS[8][16 * LDSPAD]; // per-wave [i][*]
    _Float16* pt = pS[wave];

    { // stage q tile [16 i][16 d] through per-wave LDS (same-wave DS ops in-order)
        int d  = lane & 15;
        int io = (lane >> 4) * 8;
        half8 qv = *(const half8*)(qb + (size_t)d * P + i0 + io);
#pragma unroll
        for (int u = 0; u < 8; ++u) pt[(io + u) * LDSPAD + d] = qv[u];
    }
    v16h qf = load_fragA_k16(pt, 0, LDSPAD);

    v8f o[8] = {};
    float rmax[8], rsum[8];
#pragma unroll
    for (int r = 0; r < 8; ++r) { rmax[r] = -3.0e38f; rsum[r] = 0.f; }

    for (int j0 = 0; j0 < P; j0 += 32) {
        __syncthreads();
        if (tid < 64) { // stage kS [32 j][16 d]
            int d  = tid >> 2;
            int jl = (tid & 3) * 8;
            half8 kv = *(const half8*)(kb + (size_t)d * P + j0 + jl);
#pragma unroll
            for (int u = 0; u < 8; ++u) kS[(jl + u) * 16 + d] = kv[u];
        }
        { // stage vS [128 c][32 j] (contiguous copies)
            int c  = tid >> 1;
            int jl = (tid & 1) * 16;
            const _Float16* vr = vb + (size_t)c * P + j0 + jl;
            *(half8*)(&vS[c * LDSPAD + jl])     = *(const half8*)vr;
            *(half8*)(&vS[c * LDSPAD + jl + 8]) = *(const half8*)(vr + 8);
            if (j0 + 32 < P) __builtin_prefetch(vr + 32, 0, 1);
        }
        __syncthreads();

        v8f s0 = {}, s1 = {};
        s0 = wmma_f16(qf, load_fragB_k16(kS, 0, 16), s0);
        s1 = wmma_f16(qf, load_fragB_k16(kS, 16, 16), s1);

#pragma unroll
        for (int r = 0; r < 8; ++r) { // online softmax per row (16-lane halves)
            float t = fmaxf(s0[r], s1[r]);
            t = fmaxf(t, __shfl_xor(t, 1, 32));
            t = fmaxf(t, __shfl_xor(t, 2, 32));
            t = fmaxf(t, __shfl_xor(t, 4, 32));
            t = fmaxf(t, __shfl_xor(t, 8, 32));
            float nm = fmaxf(rmax[r], t);
            float corr = __expf(rmax[r] - nm);
            rmax[r] = nm;
            float a0 = __expf(s0[r] - nm);
            float a1 = __expf(s1[r] - nm);
            float ps = a0 + a1;
            ps += __shfl_xor(ps, 1, 32);
            ps += __shfl_xor(ps, 2, 32);
            ps += __shfl_xor(ps, 4, 32);
            ps += __shfl_xor(ps, 8, 32);
            rsum[r] = rsum[r] * corr + ps;
#pragma unroll
            for (int cs = 0; cs < 8; ++cs) o[cs][r] *= corr;
            s0[r] = a0; s1[r] = a1;
        }

        { // transpose P~ (C-layout) -> A-layout via per-wave LDS
            int hi = lane >> 4, nl = lane & 15;
#pragma unroll
            for (int r = 0; r < 8; ++r) {
                int i = r + 8 * hi;
                pt[i * LDSPAD + nl]      = (_Float16)s0[r];
                pt[i * LDSPAD + 16 + nl] = (_Float16)s1[r];
            }
        }
        v16h pf = load_fragA_k32(pt, 0, LDSPAD);
        v16h vf[8];
#pragma unroll
        for (int cs = 0; cs < 8; ++cs) vf[cs] = load_fragB_k32(vS, cs * 16, LDSPAD);
#pragma unroll
        for (int cs = 0; cs < 8; ++cs) o[cs] = wmma_f16(pf, vf[cs], o[cs]);
    }

    float a = alpha[0];
    const float* fb = feat1 + (size_t)b * 128 * P;
    float* ob = out + (size_t)b * 128 * P;
    int hi = lane >> 4, nl = lane & 15;
#pragma unroll
    for (int r = 0; r < 8; ++r) {
        float inv = 1.f / rsum[r];
        int i = i0 + r + 8 * hi;
#pragma unroll
        for (int cs = 0; cs < 8; ++cs) {
            int c = cs * 16 + nl;
            size_t idx = (size_t)c * P + i;
            ob[idx] = a * (o[cs][r] * inv) + fb[idx];
        }
    }
}

// ---------------------------------------------------------------------------
// K4: channel energy = feat2 feat2^T  (128x128, K=P).  A & B frags share one tile.
// ---------------------------------------------------------------------------
__global__ __launch_bounds__(256)
void chan_energy_kernel(const float* __restrict__ feat2, float* __restrict__ energy, int P) {
    int b = blockIdx.x;
    const float* fb = feat2 + (size_t)b * 128 * P;
    float* eb = energy + (size_t)b * 128 * 128;

    __shared__ __align__(16) _Float16 fS[128 * LDSPAD];
    int tid = threadIdx.x, wave = tid >> 5, lane = tid & 31;
    int m0 = wave * 16;
    v8f acc[8] = {};

    for (int k0 = 0; k0 < P; k0 += 32) {
        __syncthreads();
        {
            int c  = tid >> 1;
            int kl = (tid & 1) * 16;
            const float* xr = fb + (size_t)c * P + k0 + kl;
#pragma unroll
            for (int u = 0; u < 16; ++u) fS[c * LDSPAD + kl + u] = (_Float16)xr[u];
            if (k0 + 32 < P) __builtin_prefetch(xr + 32, 0, 1);
        }
        __syncthreads();
        v16h af = load_fragA_k32(fS, m0, LDSPAD);
        v16h bf[8];
#pragma unroll
        for (int ns = 0; ns < 8; ++ns) bf[ns] = load_fragB_k32(fS, ns * 16, LDSPAD);
#pragma unroll
        for (int ns = 0; ns < 8; ++ns) acc[ns] = wmma_f16(af, bf[ns], acc[ns]);
    }
    int hi = lane >> 4, nl = lane & 15;
#pragma unroll
    for (int r = 0; r < 8; ++r) {
        int c = m0 + r + 8 * hi;
#pragma unroll
        for (int ns = 0; ns < 8; ++ns)
            eb[c * 128 + ns * 16 + nl] = acc[ns][r];
    }
}

// ---------------------------------------------------------------------------
// K5: attn = softmax(rowmax(E) - E) over 128, stored f16 row-major.
// ---------------------------------------------------------------------------
__global__ __launch_bounds__(256)
void chan_softmax_kernel(const float* __restrict__ energy, _Float16* __restrict__ attn) {
    int b = blockIdx.x;
    const float* eb = energy + (size_t)b * 128 * 128;
    _Float16* ab = attn + (size_t)b * 128 * 128;
    int tid = threadIdx.x, wave = tid >> 5, lane = tid & 31;
    for (int rr = 0; rr < 16; ++rr) {
        int row = wave * 16 + rr;
        const float* er = eb + row * 128;
        float v0[4], mx = -3e38f, mn = 3e38f;
#pragma unroll
        for (int u = 0; u < 4; ++u) {
            v0[u] = er[lane + u * 32];
            mx = fmaxf(mx, v0[u]);
            mn = fminf(mn, v0[u]);
        }
        for (int s = 1; s < 32; s <<= 1) {
            mx = fmaxf(mx, __shfl_xor(mx, s, 32));
            mn = fminf(mn, __shfl_xor(mn, s, 32));
        }
        float im = mx - mn; // max of (mx - e)
        float sum = 0.f, pv[4];
#pragma unroll
        for (int u = 0; u < 4; ++u) { pv[u] = __expf((mx - v0[u]) - im); sum += pv[u]; }
        for (int s = 1; s < 32; s <<= 1) sum += __shfl_xor(sum, s, 32);
        float inv = 1.f / sum;
#pragma unroll
        for (int u = 0; u < 4; ++u) ab[row * 128 + lane + u * 32] = (_Float16)(pv[u] * inv);
    }
}

// ---------------------------------------------------------------------------
// K6: sc_feat = alpha * (attn @ feat2) + feat2.  A-frags directly from f16 global.
// ---------------------------------------------------------------------------
__global__ __launch_bounds__(256)
void chan_apply_kernel(const _Float16* __restrict__ attn, const float* __restrict__ feat2,
                       const float* __restrict__ alpha, float* __restrict__ out, int P) {
    int b  = blockIdx.y;
    int p0 = blockIdx.x * 128;
    const _Float16* ab = attn + (size_t)b * 128 * 128;
    const float* fb = feat2 + (size_t)b * 128 * P;
    float* ob = out + (size_t)b * 128 * P;

    __shared__ __align__(16) _Float16 bS[128 * LDSPAD];
    int tid = threadIdx.x, wave = tid >> 5, lane = tid & 31;
    int m0 = wave * 16;
    v8f acc[8] = {};

    for (int kc = 0; kc < 4; ++kc) {
        int kb = kc * 32;
        __syncthreads();
        { // bS [128 p][32 d] from feat2[d][p]
            int d   = tid & 31;
            int pl0 = (tid >> 5) * 16;
            const float* xr = fb + (size_t)(kb + d) * P + p0 + pl0;
#pragma unroll
            for (int u = 0; u < 16; ++u) bS[(pl0 + u) * LDSPAD + d] = (_Float16)xr[u];
            if (kc < 3) __builtin_prefetch(xr + 32 * P, 0, 1);
        }
        __syncthreads();
        v16h af = load_fragA_k32(ab + kb, m0, 128);
        v16h bf[8];
#pragma unroll
        for (int ns = 0; ns < 8; ++ns) bf[ns] = load_fragB_k32(bS, ns * 16, LDSPAD);
#pragma unroll
        for (int ns = 0; ns < 8; ++ns) acc[ns] = wmma_f16(af, bf[ns], acc[ns]);
    }
    float a = alpha[0];
    int hi = lane >> 4, nl = lane & 15;
#pragma unroll
    for (int r = 0; r < 8; ++r) {
        int c = m0 + r + 8 * hi;
#pragma unroll
        for (int ns = 0; ns < 8; ++ns) {
            size_t idx = (size_t)c * P + p0 + ns * 16 + nl;
            ob[idx] = a * acc[ns][r] + fb[idx];
        }
    }
}

// ---------------------------------------------------------------------------
// K7: out = c8_w @ (sa_conv + sc_conv) + c8_b   (feat_sum add fused into staging)
// ---------------------------------------------------------------------------
__global__ __launch_bounds__(256)
void conv8_kernel(const float* __restrict__ sa, const float* __restrict__ sc,
                  const _Float16* __restrict__ wh, // [512][128] f16
                  const float* __restrict__ bias,
                  float* __restrict__ out, int P) {
    int b     = blockIdx.z;
    int m0blk = blockIdx.y * 128;
    int p0    = blockIdx.x * 128;
    const float* sab = sa + (size_t)b * 128 * P;
    const float* scb = sc + (size_t)b * 128 * P;
    float* ob = out + (size_t)b * 512 * P;

    __shared__ __align__(16) _Float16 bS[128 * LDSPAD];
    int tid = threadIdx.x, wave = tid >> 5, lane = tid & 31;
    int m0 = wave * 16;
    v8f acc[8] = {};

    for (int kc = 0; kc < 4; ++kc) {
        int kb = kc * 32;
        __syncthreads();
        {
            int d   = tid & 31;
            int pl0 = (tid >> 5) * 16;
            size_t off = (size_t)(kb + d) * P + p0 + pl0;
#pragma unroll
            for (int u = 0; u < 16; ++u)
                bS[(pl0 + u) * LDSPAD + d] = (_Float16)(sab[off + u] + scb[off + u]);
            if (kc < 3) {
                __builtin_prefetch(sab + off + 32 * P, 0, 1);
                __builtin_prefetch(scb + off + 32 * P, 0, 1);
            }
        }
        __syncthreads();
        v16h af = load_fragA_k32(wh + (size_t)m0blk * 128 + kb, m0, 128);
        v16h bf[8];
#pragma unroll
        for (int ns = 0; ns < 8; ++ns) bf[ns] = load_fragB_k32(bS, ns * 16, LDSPAD);
#pragma unroll
        for (int ns = 0; ns < 8; ++ns) acc[ns] = wmma_f16(af, bf[ns], acc[ns]);
    }
    int hi = lane >> 4, nl = lane & 15;
#pragma unroll
    for (int r = 0; r < 8; ++r) {
        int m = m0blk + m0 + r + 8 * hi;
        float bi = bias[m];
#pragma unroll
        for (int ns = 0; ns < 8; ++ns)
            ob[(size_t)m * P + p0 + ns * 16 + nl] = acc[ns][r] + bi;
    }
}

// ---------------------------------------------------------------------------
extern "C" void kernel_launch(void* const* d_in, const int* in_sizes, int n_in,
                              void* d_out, int out_size, void* d_ws, size_t ws_size,
                              hipStream_t stream) {
    (void)in_sizes; (void)n_in; (void)out_size; (void)ws_size;
    constexpr int B = 8, Cin = 512, P = 2048;

    const float* x      = (const float*)d_in[0];
    const float* c5a_w  = (const float*)d_in[1];
    const float* c5a_g  = (const float*)d_in[2];
    const float* c5a_b  = (const float*)d_in[3];
    const float* c5a_m  = (const float*)d_in[4];
    const float* c5a_v  = (const float*)d_in[5];
    const float* c5c_w  = (const float*)d_in[6];
    const float* c5c_g  = (const float*)d_in[7];
    const float* c5c_b  = (const float*)d_in[8];
    const float* c5c_m  = (const float*)d_in[9];
    const float* c5c_v  = (const float*)d_in[10];
    const float* qw     = (const float*)d_in[11];
    const float* qb     = (const float*)d_in[12];
    const float* kw     = (const float*)d_in[13];
    const float* kbias  = (const float*)d_in[14];
    const float* vw     = (const float*)d_in[15];
    const float* vb     = (const float*)d_in[16];
    const float* pa_al  = (const float*)d_in[17];
    const float* ca_al  = (const float*)d_in[18];
    const float* c51_w  = (const float*)d_in[19];
    const float* c51_g  = (const float*)d_in[20];
    const float* c51_b  = (const float*)d_in[21];
    const float* c51_m  = (const float*)d_in[22];
    const float* c51_v  = (const float*)d_in[23];
    const float* c52_w  = (const float*)d_in[24];
    const float* c52_g  = (const float*)d_in[25];
    const float* c52_b  = (const float*)d_in[26];
    const float* c52_m  = (const float*)d_in[27];
    const float* c52_v  = (const float*)d_in[28];
    const float* c8_w   = (const float*)d_in[29];
    const float* c8_b   = (const float*)d_in[30];

    char* ws = (char*)d_ws;
    size_t off = 0;
    auto carve = [&](size_t bytes) {
        void* p = ws + off;
        off += (bytes + 255) & ~(size_t)255;
        return p;
    };
    size_t fsz = (size_t)B * 128 * P * sizeof(float);
    float*    feat1   = (float*)carve(fsz);
    float*    feat2   = (float*)carve(fsz);
    float*    sa_feat = (float*)carve(fsz);
    float*    sc_feat = (float*)carve(fsz);
    float*    sa_conv = (float*)carve(fsz);
    float*    sc_conv = (float*)carve(fsz);
    _Float16* qh      = (_Float16*)carve((size_t)B * 16 * P * 2);
    _Float16* kh      = (_Float16*)carve((size_t)B * 16 * P * 2);
    _Float16* vh      = (_Float16*)carve((size_t)B * 128 * P * 2);
    float*    energy  = (float*)carve((size_t)B * 128 * 128 * 4);
    _Float16* attn    = (_Float16*)carve((size_t)B * 128 * 128 * 2);
    // repacked f16 weights
    _Float16* c5a_wh  = (_Float16*)carve((size_t)3 * 128 * Cin * 2);
    _Float16* c5c_wh  = (_Float16*)carve((size_t)3 * 128 * Cin * 2);
    _Float16* c51_wh  = (_Float16*)carve((size_t)3 * 128 * 128 * 2);
    _Float16* c52_wh  = (_Float16*)carve((size_t)3 * 128 * 128 * 2);
    _Float16* qwh     = (_Float16*)carve((size_t)16 * 128 * 2);
    _Float16* kwh     = (_Float16*)carve((size_t)16 * 128 * 2);
    _Float16* vwh     = (_Float16*)carve((size_t)128 * 128 * 2);
    _Float16* c8_wh   = (_Float16*)carve((size_t)512 * 128 * 2);

    dim3 blk(256);
    dim3 gTile(P / 128, 1, B);

    // weight repacks (tiny, once per launch)
    repack3_kernel<<<(3 * 128 * Cin + 255) / 256, blk, 0, stream>>>(c5a_w, c5a_wh, Cin);
    repack3_kernel<<<(3 * 128 * Cin + 255) / 256, blk, 0, stream>>>(c5c_w, c5c_wh, Cin);
    repack3_kernel<<<(3 * 128 * 128 + 255) / 256, blk, 0, stream>>>(c51_w, c51_wh, 128);
    repack3_kernel<<<(3 * 128 * 128 + 255) / 256, blk, 0, stream>>>(c52_w, c52_wh, 128);
    copy_h_kernel<<<(16 * 128 + 255) / 256, blk, 0, stream>>>(qw, qwh, 16 * 128);
    copy_h_kernel<<<(16 * 128 + 255) / 256, blk, 0, stream>>>(kw, kwh, 16 * 128);
    copy_h_kernel<<<(128 * 128 + 255) / 256, blk, 0, stream>>>(vw, vwh, 128 * 128);
    copy_h_kernel<<<(512 * 128 + 255) / 256, blk, 0, stream>>>(c8_w, c8_wh, 512 * 128);

    conv3_cbr_kernel<<<gTile, blk, 0, stream>>>(x, c5a_wh, c5a_g, c5a_b, c5a_m, c5a_v,
                                                feat1, Cin, P);
    conv3_cbr_kernel<<<gTile, blk, 0, stream>>>(x, c5c_wh, c5c_g, c5c_b, c5c_m, c5c_v,
                                                feat2, Cin, P);
    gemm1x1_h_kernel<<<dim3(P / 128, 1, B), blk, 0, stream>>>(feat1, qwh, qb, qh, 16, P);
    gemm1x1_h_kernel<<<dim3(P / 128, 1, B), blk, 0, stream>>>(feat1, kwh, kbias, kh, 16, P);
    gemm1x1_h_kernel<<<dim3(P / 128, 8, B), blk, 0, stream>>>(feat1, vwh, vb, vh, 128, P);
    pos_attn_kernel<<<dim3(P / 128, B), blk, 0, stream>>>(qh, kh, vh, feat1, pa_al,
                                                          sa_feat, P);
    conv3_cbr_kernel<<<gTile, blk, 0, stream>>>(sa_feat, c51_wh, c51_g, c51_b, c51_m,
                                                c51_v, sa_conv, 128, P);
    chan_energy_kernel<<<dim3(B), blk, 0, stream>>>(feat2, energy, P);
    chan_softmax_kernel<<<dim3(B), blk, 0, stream>>>(energy, attn);
    chan_apply_kernel<<<dim3(P / 128, B), blk, 0, stream>>>(attn, feat2, ca_al, sc_feat, P);
    conv3_cbr_kernel<<<gTile, blk, 0, stream>>>(sc_feat, c52_wh, c52_g, c52_b, c52_m,
                                                c52_v, sc_conv, P == 0 ? 128 : 128, P);
    conv8_kernel<<<dim3(P / 128, 4, B), blk, 0, stream>>>(sa_conv, sc_conv, c8_wh, c8_b,
                                                          (float*)d_out, P);
}
